// MultiHeadAttention_11879879544516
// MI455X (gfx1250) — matrix operand order repaired and checked by reading, not compiled, not run on hardware
//
#include <hip/hip_runtime.h>

// ---------------------------------------------------------------------------
// MultiHeadAttention for MI455X (gfx1250, wave32, WMMA).
//   1) fp32 -> bf16 converts
//   2) batched projection GEMMs (v_wmma_f32_16x16x32_bf16), K stored transposed,
//      softmax scale 1/d_k folded into Q projection
//   3) flash attention: K^T/V tiles staged in LDS via global_load_async_to_lds
//      (ASYNCcnt), online softmax, P re-enters WMMA through per-wave LDS tile
//   4) final GEMM A[4096x2048] x W_out[2048x2048] -> fp32 out
// Workspace use: 128 MiB.
// ---------------------------------------------------------------------------

typedef __attribute__((ext_vector_type(16))) __bf16 bf16x16;
typedef __attribute__((ext_vector_type(8)))  __bf16 bf16x8;
typedef __attribute__((ext_vector_type(4)))  __bf16 bf16x4;
typedef __attribute__((ext_vector_type(8)))  float  f32x8;

#define SEQ 4096
#define DA  2048
#define NH  16
#define DKH 128
#define DVH 128

union AFrag { bf16x16 v; bf16x8 h[2]; };

static __device__ inline f32x8 wmma_bf16(const AFrag& a, const AFrag& b, f32x8 c) {
  return __builtin_amdgcn_wmma_f32_16x16x32_bf16(false, a.v, false, b.v,
                                                 (short)0, c, false, false);
}

static __device__ inline float redmax16(float x) {
#pragma unroll
  for (int m = 8; m >= 1; m >>= 1) x = fmaxf(x, __shfl_xor(x, m, 32));
  return x;  // reduces within each 16-lane half (the rows that half owns)
}
static __device__ inline float redsum16(float x) {
#pragma unroll
  for (int m = 8; m >= 1; m >>= 1) x += __shfl_xor(x, m, 32);
  return x;
}

// Async global -> LDS 16-byte copy (ASYNCcnt-tracked, gfx1250).
// lds_off: per-lane byte offset into the block's LDS allocation.
static __device__ inline void async_b128(unsigned int lds_off, unsigned long long gaddr) {
  asm volatile("global_load_async_to_lds_b128 %0, %1, off"
               :: "v"(lds_off), "v"(gaddr) : "memory");
}
static __device__ inline void wait_async0() {
  asm volatile("s_wait_asynccnt 0" ::: "memory");
}
static __device__ inline unsigned int lds_off32(const void* p) {
  return (unsigned int)(unsigned long long)p;  // generic LDS ptr: low 32 bits = LDS offset
}

// ---------------------------------------------------------------------------
// fp32 -> bf16 convert, vectorized x4
// ---------------------------------------------------------------------------
__global__ __launch_bounds__(256)
void cvt_f32_bf16(const float* __restrict__ x, __bf16* __restrict__ y, int n4) {
  int i = blockIdx.x * 256 + threadIdx.x;
  if (i >= n4) return;
  float4 f = reinterpret_cast<const float4*>(x)[i];
  bf16x4 r;
  r[0] = (__bf16)f.x; r[1] = (__bf16)f.y; r[2] = (__bf16)f.z; r[3] = (__bf16)f.w;
  reinterpret_cast<bf16x4*>(y)[i] = r;
}

// ---------------------------------------------------------------------------
// Batched bf16 GEMM: C[z] = alpha * A[z] (MxK row-major) * B[z] (KxN row-major)
// CMODE 0: C fp32 row-major   1: C bf16 row-major   2: C bf16 transposed (C[n*M+m])
// 128 threads = 4 waves; each wave computes a 64x64 tile (4 A-frags x 4 B-frags
// -> 16 wmma, 0.5 fragment loads per wmma). Block tile = 256x64.
// ---------------------------------------------------------------------------
template <int CMODE>
__global__ __launch_bounds__(128)
void wmma_gemm(const __bf16* __restrict__ A, const __bf16* __restrict__ B,
               void* __restrict__ Cout, int M, int N, int K,
               long sA, long sB, long sC, float alpha) {
  const int lane = threadIdx.x & 31;
  const int wave = threadIdx.x >> 5;
  const int l16  = lane & 15;
  const int half = lane >> 4;
  const int z    = blockIdx.z;
  A += (long)z * sA;
  B += (long)z * sB;
  const int m0 = blockIdx.x * 256 + wave * 64;
  const int n0 = blockIdx.y * 64;

  f32x8 acc[4][4] = {};
  for (int kc = 0; kc < K; kc += 32) {
    AFrag a[4];
#pragma unroll
    for (int mf = 0; mf < 4; ++mf) {
      const __bf16* pa = A + (long)(m0 + mf * 16 + l16) * K + kc + half * 8;
      a[mf].h[0] = *reinterpret_cast<const bf16x8*>(pa);       // K +0..7  / +8..15
      a[mf].h[1] = *reinterpret_cast<const bf16x8*>(pa + 16);  // K +16..23 / +24..31
    }
#pragma unroll
    for (int nt = 0; nt < 4; ++nt) {
      AFrag b;  // lane = N-col, 16 K in table order
      const __bf16* pb = B + (long)(kc + half * 8) * N + n0 + nt * 16 + l16;
      for (int s = 0; s < 16; ++s)
        b.v[s] = pb[(long)((s & 7) + (s >> 3) * 16) * N];
#pragma unroll
      for (int mf = 0; mf < 4; ++mf)
        acc[mf][nt] = wmma_bf16(a[mf], b, acc[mf][nt]);
    }
  }

#pragma unroll
  for (int mf = 0; mf < 4; ++mf)
#pragma unroll
    for (int nt = 0; nt < 4; ++nt)
#pragma unroll
      for (int v = 0; v < 8; ++v) {
        const int row = m0 + mf * 16 + v + half * 8;  // C layout: vgpr v -> M (+8 hi half)
        const int col = n0 + nt * 16 + l16;           // lane -> N
        const float val = acc[mf][nt][v] * alpha;
        if (CMODE == 0)
          reinterpret_cast<float*>(Cout)[(long)z * sC + (long)row * N + col] = val;
        else if (CMODE == 1)
          reinterpret_cast<__bf16*>(Cout)[(long)z * sC + (long)row * N + col] = (__bf16)val;
        else
          reinterpret_cast<__bf16*>(Cout)[(long)z * sC + (long)col * M + row] = (__bf16)val;
      }
}

// ---------------------------------------------------------------------------
// Flash attention. blockIdx.y = head; 256 threads = 8 waves; each wave owns
// 32 query rows (mf=2) -> 256 query rows per block. Key loop in steps of 64:
//   - all threads async-copy the K^T 128x64 and V 64x128 bf16 tiles to LDS
//     (once per block instead of once per wave: 8x less global traffic)
//   - S = Q*K^T from LDS B-frags (each B-frag feeds 2 wmma)
//   - online softmax (row stats live in exactly the lanes that own the rows)
//   - P -> per-wave LDS tile -> WMMA A-operand for P*V
// ---------------------------------------------------------------------------
__global__ __launch_bounds__(256)
void flash_attn(const __bf16* __restrict__ Qp,   // [NH][SEQ][DKH] (pre-scaled 1/d_k)
                const __bf16* __restrict__ Ktp,  // [NH][DKH][SEQ]
                const __bf16* __restrict__ Vp,   // [NH][SEQ][DVH]
                __bf16* __restrict__ Abuf) {     // [SEQ][NH*DVH]
  const int tid  = threadIdx.x;
  const int lane = tid & 31;
  const int wave = tid >> 5;
  const int l16  = lane & 15;
  const int half = lane >> 4;
  const int h    = blockIdx.y;

  const __bf16* Q  = Qp  + (long)h * SEQ * DKH;
  const __bf16* Kt = Ktp + (long)h * DKH * SEQ;
  const __bf16* V  = Vp  + (long)h * SEQ * DVH;
  const int q0 = blockIdx.x * 256 + wave * 32;

  __shared__ __bf16 KtTile[128 * 64];   // 16 KB: [dk rows][64 keys]
  __shared__ __bf16 VTile[64 * 128];    // 16 KB: [64 keys][v dims]
  __shared__ __bf16 Plds[8][32 * 64];   // 32 KB: per-wave 32x64 P tile
  __bf16* pl = &Plds[wave][0];
  const unsigned int kt_base = lds_off32(&KtTile[0]);
  const unsigned int v_base  = lds_off32(&VTile[0]);

  // Q fragments: 32 rows x 128 dk, 2 row-frags x 4 k-chunks
  AFrag qf[2][4];
#pragma unroll
  for (int mf = 0; mf < 2; ++mf)
#pragma unroll
    for (int kc = 0; kc < 4; ++kc) {
      const __bf16* pq = Q + (long)(q0 + mf * 16 + l16) * DKH + kc * 32 + half * 8;
      qf[mf][kc].h[0] = *reinterpret_cast<const bf16x8*>(pq);
      qf[mf][kc].h[1] = *reinterpret_cast<const bf16x8*>(pq + 16);
    }

  f32x8 o[2][8] = {};
  float mrow[2][8], lrow[2][8];
#pragma unroll
  for (int mf = 0; mf < 2; ++mf)
#pragma unroll
    for (int v = 0; v < 8; ++v) { mrow[mf][v] = -3.0e38f; lrow[mf][v] = 0.0f; }

  for (int s0 = 0; s0 < SEQ; s0 += 64) {
    // ---- Stage K^T (128x64) and V (64x128) tiles into LDS, 16B chunks ----
    // 1024 chunks each; 256 threads x 4 chunks per tile.
#pragma unroll
    for (int i = 0; i < 4; ++i) {
      const int c   = tid + i * 256;          // 0..1023
      const int kr  = c >> 3, kcol = c & 7;   // K^T: 8 chunks per 64-col row
      async_b128(kt_base + (unsigned int)((kr * 64 + kcol * 8) * 2),
                 (unsigned long long)(const void*)(Kt + (long)kr * SEQ + s0 + kcol * 8));
      const int vr  = c >> 4, vcol = c & 15;  // V: 16 chunks per 128-col row
      async_b128(v_base + (unsigned int)((vr * 128 + vcol * 8) * 2),
                 (unsigned long long)(const void*)(V + (long)(s0 + vr) * DVH + vcol * 8));
    }
    wait_async0();
    __syncthreads();

    // ---- S = Q * K^T : 32 q-rows x 64 keys ----
    f32x8 sacc[2][4] = {};
#pragma unroll
    for (int nt = 0; nt < 4; ++nt) {
#pragma unroll
      for (int kc = 0; kc < 4; ++kc) {
        AFrag b;  // lane = key column of K^T tile, 16 dk rows in table order
        const __bf16* pb = &KtTile[(kc * 32 + half * 8) * 64 + nt * 16 + l16];
        for (int s = 0; s < 16; ++s)
          b.v[s] = pb[((s & 7) + (s >> 3) * 16) * 64];
#pragma unroll
        for (int mf = 0; mf < 2; ++mf)
          sacc[mf][nt] = wmma_bf16(qf[mf][kc], b, sacc[mf][nt]);
      }
    }

    // ---- Online softmax per 16-row fragment ----
#pragma unroll
    for (int mf = 0; mf < 2; ++mf) {
      float mnew[8], corr[8], rs[8];
#pragma unroll
      for (int v = 0; v < 8; ++v) {
        float mx = fmaxf(fmaxf(sacc[mf][0][v], sacc[mf][1][v]),
                         fmaxf(sacc[mf][2][v], sacc[mf][3][v]));
        mx = redmax16(mx);
        mnew[v] = fmaxf(mrow[mf][v], mx);
        corr[v] = __expf(mrow[mf][v] - mnew[v]);
        rs[v] = 0.0f;
      }
#pragma unroll
      for (int nt = 0; nt < 4; ++nt)
#pragma unroll
        for (int v = 0; v < 8; ++v) {
          float p = __expf(sacc[mf][nt][v] - mnew[v]);
          sacc[mf][nt][v] = p;
          rs[v] += p;
        }
#pragma unroll
      for (int v = 0; v < 8; ++v) {
        rs[v] = redsum16(rs[v]);
        lrow[mf][v] = lrow[mf][v] * corr[v] + rs[v];
        mrow[mf][v] = mnew[v];
      }
#pragma unroll
      for (int nt = 0; nt < 8; ++nt)
#pragma unroll
        for (int v = 0; v < 8; ++v) o[mf][nt][v] *= corr[v];

      // P (C-layout) -> row-major rows [mf*16 .. mf*16+15] of this wave's tile
#pragma unroll
      for (int nt = 0; nt < 4; ++nt)
#pragma unroll
        for (int v = 0; v < 8; ++v)
          pl[(mf * 16 + v + half * 8) * 64 + nt * 16 + l16] = (__bf16)sacc[mf][nt][v];
    }
    __asm__ volatile("s_wait_dscnt 0" ::: "memory");  // same-wave LDS RAW fence

    // ---- O += P(32x64) * V(64x128) ----
#pragma unroll
    for (int kc2 = 0; kc2 < 2; ++kc2) {
      AFrag pa[2];
#pragma unroll
      for (int mf = 0; mf < 2; ++mf) {
        const __bf16* pp = pl + (mf * 16 + l16) * 64 + kc2 * 32 + half * 8;
        pa[mf].h[0] = *reinterpret_cast<const bf16x8*>(pp);
        pa[mf].h[1] = *reinterpret_cast<const bf16x8*>(pp + 16);
      }
#pragma unroll
      for (int nt = 0; nt < 8; ++nt) {
        AFrag b;  // lane = v-dim column of V tile, 16 key rows in table order
        const __bf16* pv = &VTile[(kc2 * 32 + half * 8) * 128 + nt * 16 + l16];
        for (int s = 0; s < 16; ++s)
          b.v[s] = pv[((s & 7) + (s >> 3) * 16) * 128];
#pragma unroll
        for (int mf = 0; mf < 2; ++mf)
          o[mf][nt] = wmma_bf16(pa[mf], b, o[mf][nt]);
      }
    }
    __syncthreads();  // protect K^T/V tiles from next iteration's async overwrite
  }

  // ---- Normalize, write concat-head A (bf16) ----
#pragma unroll
  for (int mf = 0; mf < 2; ++mf) {
#pragma unroll
    for (int v = 0; v < 8; ++v) lrow[mf][v] = 1.0f / lrow[mf][v];
#pragma unroll
    for (int nt = 0; nt < 8; ++nt)
#pragma unroll
      for (int v = 0; v < 8; ++v) {
        const int row = q0 + mf * 16 + v + half * 8;
        const int col = h * DVH + nt * 16 + l16;
        Abuf[(long)row * DA + col] = (__bf16)(o[mf][nt][v] * lrow[mf][v]);
      }
  }
}

// ---------------------------------------------------------------------------
extern "C" void kernel_launch(void* const* d_in, const int* in_sizes, int n_in,
                              void* d_out, int out_size, void* d_ws, size_t ws_size,
                              hipStream_t stream) {
  const float* q   = (const float*)d_in[0];
  const float* key = (const float*)d_in[1];
  const float* val = (const float*)d_in[2];
  const float* Eq  = (const float*)d_in[3];
  const float* Ek  = (const float*)d_in[4];
  const float* Ev  = (const float*)d_in[5];
  const float* Wo  = (const float*)d_in[6];
  float* out = (float*)d_out;

  // Workspace carve-up (total 128 MiB)
  char* ws = (char*)d_ws;
  size_t off = 0;
  auto take = [&](size_t elems) {
    __bf16* p = (__bf16*)(ws + off);
    off += elems * sizeof(__bf16);
    return p;
  };
  __bf16* qb  = take((size_t)SEQ * DA);
  __bf16* kb  = take((size_t)SEQ * DA);
  __bf16* vb  = take((size_t)SEQ * DA);
  __bf16* Eqb = take((size_t)NH * DA * DKH);
  __bf16* Ekb = take((size_t)NH * DA * DKH);
  __bf16* Evb = take((size_t)NH * DA * DVH);
  __bf16* Wob = take((size_t)DA * DA);
  __bf16* Qp  = take((size_t)NH * SEQ * DKH);
  __bf16* Ktp = take((size_t)NH * DKH * SEQ);
  __bf16* Vp  = take((size_t)NH * SEQ * DVH);
  __bf16* Abuf = qb;  // queries-bf16 dead after projections; reuse for concat-A

  auto cvt = [&](const float* src, __bf16* dst, long n) {
    int n4 = (int)(n / 4);
    cvt_f32_bf16<<<(n4 + 255) / 256, 256, 0, stream>>>(src, dst, n4);
  };
  cvt(q,   qb,  (long)SEQ * DA);
  cvt(key, kb,  (long)SEQ * DA);
  cvt(val, vb,  (long)SEQ * DA);
  cvt(Eq,  Eqb, (long)NH * DA * DKH);
  cvt(Ek,  Ekb, (long)NH * DA * DKH);
  cvt(Ev,  Evb, (long)NH * DA * DVH);
  cvt(Wo,  Wob, (long)DA * DA);

  // Projections (batched over heads). Fold softmax scale 1/d_k into Q.
  const float qscale = 1.0f / (float)DKH;
  wmma_gemm<1><<<dim3(SEQ / 256, DKH / 64, NH), 128, 0, stream>>>(
      qb, Eqb, Qp, SEQ, DKH, DA, 0, (long)DA * DKH, (long)SEQ * DKH, qscale);
  wmma_gemm<2><<<dim3(SEQ / 256, DKH / 64, NH), 128, 0, stream>>>(
      kb, Ekb, Ktp, SEQ, DKH, DA, 0, (long)DA * DKH, (long)DKH * SEQ, 1.0f);
  wmma_gemm<1><<<dim3(SEQ / 256, DVH / 64, NH), 128, 0, stream>>>(
      vb, Evb, Vp, SEQ, DVH, DA, 0, (long)DA * DVH, (long)SEQ * DVH, 1.0f);

  // Flash attention -> concat-head A (bf16)
  flash_attn<<<dim3(SEQ / 256, NH), 256, 0, stream>>>(Qp, Ktp, Vp, Abuf);

  // Output projection -> fp32
  wmma_gemm<0><<<dim3(SEQ / 256, DA / 64, 1), 128, 0, stream>>>(
      Abuf, Wob, out, SEQ, DA, DA, 0, 0, 0, 1.0f);
}
